// MambaSWELU_1520418423437
// MI455X (gfx1250) — compile-verified, matrix operand order checked by reading, not run on hardware
//
#include <hip/hip_runtime.h>
#include <hip/hip_bf16.h>

// ---------------- model constants (match reference) ----------------
constexpr int V_  = 50257;
constexpr int D_  = 1024;
constexpr int L_  = 4;
constexpr int N_  = 16;
constexpr int KC_ = 4;
constexpr int DI_ = 2048;
constexpr int S_  = 512;
constexpr int B_  = 2;
constexpr int H_  = 2048;
constexpr int R_  = 64;
constexpr int M_  = B_ * S_;          // 1024 rows for every GEMM

typedef __attribute__((ext_vector_type(16))) __bf16 v16bf;
typedef __attribute__((ext_vector_type(8)))  __bf16 v8bf;
typedef __attribute__((ext_vector_type(8)))  float  v8f;

__device__ __forceinline__ __bf16 to_bf16(float f){
  union { float f; unsigned u; } v; v.f = f;
  unsigned r = v.u + 0x7fffu + ((v.u >> 16) & 1u);   // round-to-nearest-even
  union { unsigned short s; __bf16 b; } o; o.s = (unsigned short)(r >> 16);
  return o.b;
}
__device__ __forceinline__ float sweluf(float x, float k){
  return x / (1.f + __expf(-k * x));
}
__device__ __forceinline__ float softplusf(float x){
  return x > 20.f ? x : log1pf(__expf(x));
}

// Assemble a 16x32 bf16 WMMA fragment row for this lane.
// Per ISA layout: lanes 0-15 carry K = [k0..k0+7, k0+16..k0+23],
// lanes 16-31 carry K = [k0+8..k0+15, k0+24..k0+31] -> two 16B loads.
__device__ __forceinline__ v16bf load_frag(const __bf16* __restrict__ row, int kA, int kB){
  v8bf lo = *reinterpret_cast<const v8bf*>(row + kA);
  v8bf hi = *reinterpret_cast<const v8bf*>(row + kB);
  v16bf r;
#pragma unroll
  for (int i = 0; i < 8; ++i){ r[i] = lo[i]; r[i + 8] = hi[i]; }
  return r;
}

__device__ __forceinline__ v16bf zero_frag(){
  v16bf r;
#pragma unroll
  for (int i = 0; i < 16; ++i) r[i] = to_bf16(0.f);
  return r;
}

#define WMMA_BF16(Af, Bf, Cf) \
  __builtin_amdgcn_wmma_f32_16x16x32_bf16(false, (Af), false, (Bf), (short)0, (Cf), false, false)

// ---------------- generic bf16 WMMA GEMM ----------------
// C[M,N] = A[M,K](bf16,row-major) * Bt[N,K](bf16, i.e. B transposed) (+ Cin)
// 8 waves/block; each wave computes one 16(M) x 64(N) strip with 4 WMMAs/k-step.
// Interior waves (full 64-col strip in bounds, K multiple of 64) run a
// branch-free 2-stage software pipeline: stage-0 fragments are loop-carried
// registers written directly by loads (no register rotation copies), stage-1
// fragments are loop-locals; the last iteration is peeled.
__global__ __launch_bounds__(256) void gemm_bf16_wmma(
    const __bf16* __restrict__ A, const __bf16* __restrict__ Bt,
    const float* __restrict__ Cin, float* __restrict__ C,
    int M, int N, int K)
{
  const int lane = threadIdx.x & 31;
  const int wave = threadIdx.x >> 5;                 // 0..7
  const int m0 = blockIdx.y * 32 + (wave >> 2) * 16; // 2 waves along M
  const int n0 = blockIdx.x * 256 + (wave & 3) * 64; // 4 waves along N
  if (m0 >= M || n0 >= N) return;                    // wave-uniform
  const int lm   = lane & 15;
  const int half = lane >> 4;

  v8f acc[4];
#pragma unroll
  for (int j = 0; j < 4; ++j){
    int n = n0 + j * 16 + lm;
#pragma unroll
    for (int r = 0; r < 8; ++r){
      int m = m0 + half * 8 + r;                     // C layout: VGPR r, halves at M+8
      acc[j][r] = (Cin && n < N) ? Cin[(size_t)m * N + n] : 0.f;
    }
  }

  const __bf16* arow = A + (size_t)(m0 + lm) * K;
  const int kA0 = half * 8, kB0 = 16 + half * 8;

  if (n0 + 64 <= N && (K & 63) == 0){
    // ---- fast interior path: branch-free 2-stage pipeline, unroll x2 ----
    const __bf16* br0 = Bt + (size_t)(n0 +  0 + lm) * K;
    const __bf16* br1 = Bt + (size_t)(n0 + 16 + lm) * K;
    const __bf16* br2 = Bt + (size_t)(n0 + 32 + lm) * K;
    const __bf16* br3 = Bt + (size_t)(n0 + 48 + lm) * K;

    // stage-0 fragments (loop-carried; always written by loads, never copied)
    v16bf af0 = load_frag(arow, kA0, kB0);
    v16bf b00 = load_frag(br0,  kA0, kB0);
    v16bf b10 = load_frag(br1,  kA0, kB0);
    v16bf b20 = load_frag(br2,  kA0, kB0);
    v16bf b30 = load_frag(br3,  kA0, kB0);

    int k0 = 0;
    for (; k0 + 64 < K; k0 += 64){
      const int k1 = k0 + 32;
      v16bf af1 = load_frag(arow, k1 + kA0, k1 + kB0);
      v16bf b01 = load_frag(br0,  k1 + kA0, k1 + kB0);
      v16bf b11 = load_frag(br1,  k1 + kA0, k1 + kB0);
      v16bf b21 = load_frag(br2,  k1 + kA0, k1 + kB0);
      v16bf b31 = load_frag(br3,  k1 + kA0, k1 + kB0);
      __builtin_prefetch(arow + k0 + 64, 0, 3);

      acc[0] = WMMA_BF16(af0, b00, acc[0]);
      acc[1] = WMMA_BF16(af0, b10, acc[1]);
      acc[2] = WMMA_BF16(af0, b20, acc[2]);
      acc[3] = WMMA_BF16(af0, b30, acc[3]);

      const int k2 = k0 + 64;
      af0 = load_frag(arow, k2 + kA0, k2 + kB0);
      b00 = load_frag(br0,  k2 + kA0, k2 + kB0);
      b10 = load_frag(br1,  k2 + kA0, k2 + kB0);
      b20 = load_frag(br2,  k2 + kA0, k2 + kB0);
      b30 = load_frag(br3,  k2 + kA0, k2 + kB0);

      acc[0] = WMMA_BF16(af1, b01, acc[0]);
      acc[1] = WMMA_BF16(af1, b11, acc[1]);
      acc[2] = WMMA_BF16(af1, b21, acc[2]);
      acc[3] = WMMA_BF16(af1, b31, acc[3]);
    }
    // peeled tail: k0 == K-64, stages 0 and 1 only (no further loads)
    {
      const int k1 = k0 + 32;
      v16bf af1 = load_frag(arow, k1 + kA0, k1 + kB0);
      v16bf b01 = load_frag(br0,  k1 + kA0, k1 + kB0);
      v16bf b11 = load_frag(br1,  k1 + kA0, k1 + kB0);
      v16bf b21 = load_frag(br2,  k1 + kA0, k1 + kB0);
      v16bf b31 = load_frag(br3,  k1 + kA0, k1 + kB0);

      acc[0] = WMMA_BF16(af0, b00, acc[0]);
      acc[1] = WMMA_BF16(af0, b10, acc[1]);
      acc[2] = WMMA_BF16(af0, b20, acc[2]);
      acc[3] = WMMA_BF16(af0, b30, acc[3]);

      acc[0] = WMMA_BF16(af1, b01, acc[0]);
      acc[1] = WMMA_BF16(af1, b11, acc[1]);
      acc[2] = WMMA_BF16(af1, b21, acc[2]);
      acc[3] = WMMA_BF16(af1, b31, acc[3]);
    }
  } else {
    // ---- N-edge / irregular-K path: per-subtile guards ----
    for (int k0 = 0; k0 < K; k0 += 32){
      v16bf af = load_frag(arow, k0 + kA0, k0 + kB0);
#pragma unroll
      for (int j = 0; j < 4; ++j){
        int n = n0 + j * 16 + lm;
        v16bf bfv = (n < N) ? load_frag(Bt + (size_t)n * K, k0 + kA0, k0 + kB0)
                            : zero_frag();
        acc[j] = WMMA_BF16(af, bfv, acc[j]);
      }
    }
  }

#pragma unroll
  for (int j = 0; j < 4; ++j){
    int n = n0 + j * 16 + lm;
    if (n >= N) continue;
#pragma unroll
    for (int r = 0; r < 8; ++r){
      int m = m0 + half * 8 + r;
      C[(size_t)m * N + n] = acc[j][r];
    }
  }
}

// ---------------- prep kernels ----------------
__global__ __launch_bounds__(256) void convert_bf16_kernel(
    const float* __restrict__ s, __bf16* __restrict__ d, size_t n)
{
  size_t i = (size_t)blockIdx.x * 256 + threadIdx.x;
  size_t stride = (size_t)gridDim.x * 256;
  for (; i < n; i += stride) d[i] = to_bf16(s[i]);
}

// dst[n][k] = bf16(src[k][n]) : weight [K][N] -> N-major bf16 [N][K]
__global__ __launch_bounds__(256) void transpose_bf16_kernel(
    const float* __restrict__ src, __bf16* __restrict__ dst, int Kd, int Nd)
{
  __shared__ float tile[32][33];
  int nb = blockIdx.x * 32, kb = blockIdx.y * 32;
  int tx = threadIdx.x & 31, ty = threadIdx.x >> 5;  // ty 0..7
  for (int i = ty; i < 32; i += 8){
    int k = kb + i, n = nb + tx;
    tile[i][tx] = (k < Kd && n < Nd) ? src[(size_t)k * Nd + n] : 0.f;
  }
  __syncthreads();
  for (int i = ty; i < 32; i += 8){
    int n = nb + i, k = kb + tx;
    if (n < Nd && k < Kd) dst[(size_t)n * Kd + k] = to_bf16(tile[tx][i]);
  }
}

__global__ __launch_bounds__(256) void slice_bf16_kernel(
    const float* __restrict__ s, __bf16* __restrict__ d, int rows, int ld, int cols)
{
  int i = blockIdx.x * 256 + threadIdx.x;
  if (i >= rows * cols) return;
  int r = i / cols, c = i - r * cols;
  d[i] = to_bf16(s[(size_t)r * ld + c]);
}

// ---------------- forward kernels ----------------
__global__ __launch_bounds__(256) void embed_kernel(
    const float* __restrict__ tok, const float* __restrict__ pos,
    const int* __restrict__ ids, float* __restrict__ x)
{
  int idx = blockIdx.x * 256 + threadIdx.x;          // < M_*D_
  int d = idx % D_, t = idx / D_, s = t % S_;
  x[idx] = tok[(size_t)ids[t] * D_ + d] + pos[(size_t)s * D_ + d];
}

__global__ __launch_bounds__(256) void rmsnorm_bf16_kernel(
    const float* __restrict__ x, const float* __restrict__ w, __bf16* __restrict__ h)
{
  int r = blockIdx.x;
  const float* xr = x + (size_t)r * D_;
  __shared__ float red[256];
  float ss = 0.f;
  for (int i = threadIdx.x; i < D_; i += 256){ float v = xr[i]; ss += v * v; }
  red[threadIdx.x] = ss; __syncthreads();
  for (int o = 128; o > 0; o >>= 1){
    if (threadIdx.x < o) red[threadIdx.x] += red[threadIdx.x + o];
    __syncthreads();
  }
  float inv = rsqrtf(red[0] / (float)D_ + 1e-5f);
  for (int i = threadIdx.x; i < D_; i += 256)
    h[(size_t)r * D_ + i] = to_bf16(xr[i] * inv * w[i]);
}

// causal depthwise conv (KC taps) + bias + SWeLU; also swelu(z) for the gate
__global__ __launch_bounds__(256) void conv_swelu_kernel(
    const float* __restrict__ xz, const float* __restrict__ cw,
    const float* __restrict__ cb, const float* __restrict__ kptr,
    float* __restrict__ u2f, __bf16* __restrict__ u2b, float* __restrict__ swz)
{
  int idx = blockIdx.x * 256 + threadIdx.x;          // < M_*DI_
  int d = idx % DI_;
  int t = idx / DI_;
  int s = t % S_;
  int b = t / S_;
  float k = kptr[0];
  float acc = cb[d];
#pragma unroll
  for (int kk = 0; kk < KC_; ++kk){
    int sp = s + kk - (KC_ - 1);
    if (sp >= 0)
      acc += xz[((size_t)(b * S_ + sp)) * (2 * DI_) + d] * cw[d * KC_ + kk];
  }
  float u2 = sweluf(acc, k);
  u2f[idx] = u2;
  u2b[idx] = to_bf16(u2);
  float z = xz[(size_t)t * (2 * DI_) + DI_ + d];
  swz[idx] = sweluf(z, k);
}

// selective scan: one channel per thread, B/C broadcast through LDS each step
__global__ __launch_bounds__(256) void scan_kernel(
    const float* __restrict__ dtraw, const float* __restrict__ dtb,
    const float* __restrict__ Alog,  const float* __restrict__ xdbl,
    const float* __restrict__ u2f,   const float* __restrict__ Dsk,
    const float* __restrict__ swz,   __bf16* __restrict__ ybf)
{
  int b = blockIdx.x;
  int d = blockIdx.y * 256 + threadIdx.x;
  float A[N_];
#pragma unroll
  for (int n = 0; n < N_; ++n) A[n] = -__expf(Alog[(size_t)d * N_ + n]);
  float h[N_] = {};
  float bias = dtb[d];
  float Dd = Dsk[d];
  __shared__ float sB[N_], sC[N_];
  for (int s = 0; s < S_; ++s){
    size_t t = (size_t)b * S_ + s;
    __syncthreads();
    if (threadIdx.x < 2 * N_){
      float v = xdbl[t * (R_ + 2 * N_) + R_ + threadIdx.x];
      if (threadIdx.x < N_) sB[threadIdx.x] = v; else sC[threadIdx.x - N_] = v;
    }
    __syncthreads();
    float dt = softplusf(dtraw[t * DI_ + d] + bias);
    float u  = u2f[t * DI_ + d];
    float du = dt * u;
    float y = 0.f;
#pragma unroll
    for (int n = 0; n < N_; ++n){
      h[n] = __expf(dt * A[n]) * h[n] + du * sB[n];
      y += h[n] * sC[n];
    }
    y = (y + Dd * u) * swz[t * DI_ + d];
    ybf[t * DI_ + d] = to_bf16(y);
  }
}

__global__ __launch_bounds__(256) void bias_swelu_bf16_kernel(
    const float* __restrict__ in, const float* __restrict__ bias,
    const float* __restrict__ kptr, __bf16* __restrict__ outb, int cols)
{
  int i = blockIdx.x * 256 + threadIdx.x;
  float k = kptr[0];
  float v = in[i] + bias[i % cols];
  outb[i] = to_bf16(sweluf(v, k));
}

__global__ void zero_kernel(float* p){ *p = 0.f; }

// one block per (b, s<S-1): log-softmax over V + NLL at label, mean via atomic
__global__ __launch_bounds__(256) void nll_kernel(
    const float* __restrict__ logits, const int* __restrict__ labels, float* __restrict__ loss)
{
  int row = blockIdx.x;                              // < B_*(S_-1)
  int b = row / (S_ - 1), s = row % (S_ - 1);
  const float* lr = logits + ((size_t)b * S_ + s) * V_;
  __shared__ float red[256];
  float mx = -3.4e38f;
  for (int i = threadIdx.x; i < V_; i += 256) mx = fmaxf(mx, lr[i]);
  red[threadIdx.x] = mx; __syncthreads();
  for (int o = 128; o > 0; o >>= 1){
    if (threadIdx.x < o) red[threadIdx.x] = fmaxf(red[threadIdx.x], red[threadIdx.x + o]);
    __syncthreads();
  }
  mx = red[0]; __syncthreads();
  float sm = 0.f;
  for (int i = threadIdx.x; i < V_; i += 256) sm += __expf(lr[i] - mx);
  red[threadIdx.x] = sm; __syncthreads();
  for (int o = 128; o > 0; o >>= 1){
    if (threadIdx.x < o) red[threadIdx.x] += red[threadIdx.x + o];
    __syncthreads();
  }
  if (threadIdx.x == 0){
    int lab = labels[b * S_ + s + 1];
    float nll = mx + __logf(red[0]) - lr[lab];
    atomicAdd(loss, nll * (1.f / (float)(B_ * (S_ - 1))));
  }
}

// ---------------- launch ----------------
extern "C" void kernel_launch(void* const* d_in, const int* in_sizes, int n_in,
                              void* d_out, int out_size, void* d_ws, size_t ws_size,
                              hipStream_t stream)
{
  const float* tok_emb  = (const float*)d_in[0];
  const float* pos_emb  = (const float*)d_in[1];
  const float* norm_w   = (const float*)d_in[2];
  const float* in_proj  = (const float*)d_in[3];
  const float* conv_w   = (const float*)d_in[4];
  const float* conv_b   = (const float*)d_in[5];
  const float* x_proj   = (const float*)d_in[6];
  const float* dt_proj  = (const float*)d_in[7];
  const float* dt_pb    = (const float*)d_in[8];
  const float* A_log    = (const float*)d_in[9];
  const float* D_skip   = (const float*)d_in[10];
  const float* out_proj = (const float*)d_in[11];
  const float* k_block  = (const float*)d_in[12];
  const float* d1w = (const float*)d_in[13];
  const float* d1b = (const float*)d_in[14];
  const float* d2w = (const float*)d_in[15];
  const float* d2b = (const float*)d_in[16];
  const float* d3w = (const float*)d_in[17];
  const float* d3b = (const float*)d_in[18];
  const float* ks  = (const float*)d_in[19];
  const int* ids    = (const int*)d_in[20];
  const int* labels = (const int*)d_in[21];

  float* logits = (float*)d_out;
  float* loss   = logits + (size_t)M_ * V_;

  char* base = (char*)d_ws; size_t off = 0;
  auto alloc = [&](size_t bytes) -> void* {
    off = (off + 255) & ~(size_t)255;
    void* p = base + off; off += bytes; return p;
  };
  constexpr int XD = R_ + 2 * N_;  // 96

  __bf16* tokb     = (__bf16*)alloc((size_t)V_ * D_ * 2);
  __bf16* inprojT  = (__bf16*)alloc((size_t)L_ * 2 * DI_ * D_ * 2);
  __bf16* xprojT   = (__bf16*)alloc((size_t)L_ * XD * DI_ * 2);
  __bf16* dtprojT  = (__bf16*)alloc((size_t)L_ * DI_ * R_ * 2);
  __bf16* outprojT = (__bf16*)alloc((size_t)L_ * D_ * DI_ * 2);
  __bf16* d1T = (__bf16*)alloc((size_t)H_ * D_ * 2);
  __bf16* d2T = (__bf16*)alloc((size_t)H_ * H_ * 2);
  __bf16* d3T = (__bf16*)alloc((size_t)D_ * H_ * 2);
  float*  x    = (float*)alloc((size_t)M_ * D_ * 4);
  __bf16* hbf  = (__bf16*)alloc((size_t)M_ * D_ * 2);
  float*  xz   = (float*)alloc((size_t)M_ * 2 * DI_ * 4);
  float*  u2f  = (float*)alloc((size_t)M_ * DI_ * 4);
  __bf16* u2b  = (__bf16*)alloc((size_t)M_ * DI_ * 2);
  float*  swz  = (float*)alloc((size_t)M_ * DI_ * 4);
  float*  xdbl = (float*)alloc((size_t)M_ * XD * 4);
  __bf16* dtrb = (__bf16*)alloc((size_t)M_ * R_ * 2);
  float*  dtraw= (float*)alloc((size_t)M_ * DI_ * 4);
  __bf16* ybf  = (__bf16*)alloc((size_t)M_ * DI_ * 2);

  // buffer reuse for the MLP tail (layer scratch is dead by then)
  float*  t1  = xz;        // M*H fp32 fits in M*2DI
  __bf16* t1b = u2b;
  float*  t2  = dtraw;
  __bf16* t2b = ybf;
  float*  t3  = x;         // x consumed into bf16 before t3 is written
  __bf16* xbf = hbf;

  dim3 blk(256);
  auto cdiv = [](int a, int b){ return (a + b - 1) / b; };

  // ---- weight prep: bf16 convert + N-major transposes ----
  convert_bf16_kernel<<<4096, blk, 0, stream>>>(tok_emb, tokb, (size_t)V_ * D_);
  auto transpose = [&](const float* src, __bf16* dst, int Kd, int Nd){
    dim3 g(cdiv(Nd, 32), cdiv(Kd, 32));
    transpose_bf16_kernel<<<g, blk, 0, stream>>>(src, dst, Kd, Nd);
  };
  for (int l = 0; l < L_; ++l){
    transpose(in_proj  + (size_t)l * D_ * 2 * DI_, inprojT  + (size_t)l * 2 * DI_ * D_, D_,  2 * DI_);
    transpose(x_proj   + (size_t)l * DI_ * XD,     xprojT   + (size_t)l * XD * DI_,     DI_, XD);
    transpose(dt_proj  + (size_t)l * R_ * DI_,     dtprojT  + (size_t)l * DI_ * R_,     R_,  DI_);
    transpose(out_proj + (size_t)l * DI_ * D_,     outprojT + (size_t)l * D_ * DI_,     DI_, D_);
  }
  transpose(d1w, d1T, D_, H_);
  transpose(d2w, d2T, H_, H_);
  transpose(d3w, d3T, H_, D_);

  auto gemm = [&](const __bf16* A, const __bf16* Bt, const float* Cin, float* C,
                  int Mm, int Nn, int Kk){
    dim3 g(cdiv(Nn, 256), Mm / 32);
    gemm_bf16_wmma<<<g, blk, 0, stream>>>(A, Bt, Cin, C, Mm, Nn, Kk);
  };

  // ---- embedding ----
  embed_kernel<<<(M_ * D_) / 256, blk, 0, stream>>>(tok_emb, pos_emb, ids, x);

  // ---- Mamba layers ----
  for (int l = 0; l < L_; ++l){
    rmsnorm_bf16_kernel<<<M_, blk, 0, stream>>>(x, norm_w + (size_t)l * D_, hbf);
    gemm(hbf, inprojT + (size_t)l * 2 * DI_ * D_, nullptr, xz, M_, 2 * DI_, D_);
    conv_swelu_kernel<<<(M_ * DI_) / 256, blk, 0, stream>>>(
        xz, conv_w + (size_t)l * DI_ * KC_, conv_b + (size_t)l * DI_,
        k_block + l, u2f, u2b, swz);
    gemm(u2b, xprojT + (size_t)l * XD * DI_, nullptr, xdbl, M_, XD, DI_);
    slice_bf16_kernel<<<cdiv(M_ * R_, 256), blk, 0, stream>>>(xdbl, dtrb, M_, XD, R_);
    gemm(dtrb, dtprojT + (size_t)l * DI_ * R_, nullptr, dtraw, M_, DI_, R_);
    scan_kernel<<<dim3(B_, DI_ / 256), blk, 0, stream>>>(
        dtraw, dt_pb + (size_t)l * DI_, A_log + (size_t)l * DI_ * N_,
        xdbl, u2f, D_skip + (size_t)l * DI_, swz, ybf);
    gemm(ybf, outprojT + (size_t)l * D_ * DI_, x, x, M_, D_, DI_);  // fused residual
  }

  // ---- MLP head ----
  convert_bf16_kernel<<<1024, blk, 0, stream>>>(x, xbf, (size_t)M_ * D_);
  gemm(xbf, d1T, nullptr, t1, M_, H_, D_);
  bias_swelu_bf16_kernel<<<(M_ * H_) / 256, blk, 0, stream>>>(t1, d1b, ks + 0, t1b, H_);
  gemm(t1b, d2T, nullptr, t2, M_, H_, H_);
  bias_swelu_bf16_kernel<<<(M_ * H_) / 256, blk, 0, stream>>>(t2, d2b, ks + 1, t2b, H_);
  gemm(t2b, d3T, nullptr, t3, M_, D_, H_);
  bias_swelu_bf16_kernel<<<(M_ * D_) / 256, blk, 0, stream>>>(t3, d3b, ks + 2, xbf, D_);

  // ---- logits (dominant GEMM: 1024 x 50257 x 1024) ----
  gemm(xbf, tokb, nullptr, logits, M_, V_, D_);

  // ---- loss ----
  zero_kernel<<<1, 1, 0, stream>>>(loss);
  nll_kernel<<<B_ * (S_ - 1), blk, 0, stream>>>(logits, labels, loss);
}
